// WindowAttention_8108898255016
// MI455X (gfx1250) — matrix operand order repaired and checked by reading, not compiled, not run on hardware
//
#include <hip/hip_runtime.h>
#include <hip/hip_bf16.h>

typedef __attribute__((ext_vector_type(16))) _Float16 v16h;
typedef __attribute__((ext_vector_type(8)))  _Float16 v8h;
typedef __attribute__((ext_vector_type(8)))  float    v8f;
typedef __attribute__((ext_vector_type(4)))  int      v4i;

#define B_SZ   256
#define NTOK   196
#define DIMC   384
#define HEADS  12
#define HD     32
#define NWIN   64
#define NPAD   224          // 14 tiles of 16; 7 k-steps of 32 for PV
#define SCALEF 0.17677669529663687f

#if defined(__has_builtin)
#  if __has_builtin(__builtin_amdgcn_global_load_async_to_lds_b128) && \
      __has_builtin(__builtin_amdgcn_s_wait_asynccnt)
#    define USE_ASYNC_LDS 1
#  endif
#endif
#ifndef USE_ASYNC_LDS
#  define USE_ASYNC_LDS 0
#endif

typedef __attribute__((address_space(1))) v4i gv4i;  // global int4
typedef __attribute__((address_space(3))) v4i lv4i;  // LDS int4

// Stage 16 bytes global(f16) -> LDS. Async DMA path on CDNA5 (ASYNCcnt),
// fallback: b128 load + ds_store_b128.
static __device__ __forceinline__ void stage16(const _Float16* __restrict__ g,
                                               _Float16* l) {
#if USE_ASYNC_LDS
  __builtin_amdgcn_global_load_async_to_lds_b128((gv4i*)g, (lv4i*)l, 0, 0);
#else
  *reinterpret_cast<v8h*>(l) = *reinterpret_cast<const v8h*>(g);
#endif
}
static __device__ __forceinline__ void stage_wait() {
#if USE_ASYNC_LDS
  __builtin_amdgcn_s_wait_asynccnt(0);
#endif
  __syncthreads();
}

// Build a 16x32 f16 A-fragment (v16h), contiguous k order (matches B loads):
// lanes 0-15 : M=lane,    elems 0-15 = K0..15
// lanes16-31 : M=lane-16, elems 0-15 = K16..31
static __device__ __forceinline__ v16h make_a_frag(const _Float16* row, int koff) {
  v8h lo = *reinterpret_cast<const v8h*>(row + 2 * koff);
  v8h hi = *reinterpret_cast<const v8h*>(row + 2 * koff + 8);
  v16h a;
#pragma unroll
  for (int i = 0; i < 8; ++i) { a[i] = lo[i]; a[i + 8] = hi[i]; }
  return a;
}

// ---------------- generic f32 -> f16 conversion
__global__ __launch_bounds__(256) void k_cvt(const float* __restrict__ src,
                                             _Float16* __restrict__ dst, int n) {
  int i = blockIdx.x * 256 + threadIdx.x;
  if (i < n) dst[i] = (_Float16)src[i];
}

// ---------------- bias table expansion: (1183,12) -> (12,224,224), pad = -1e9
__global__ __launch_bounds__(256) void k_bias(const float* __restrict__ table,
                                              float* __restrict__ bias224) {
  int idx = blockIdx.x * 256 + threadIdx.x;
  const int total = HEADS * NPAD * NPAD;
  if (idx >= total) return;
  int h = idx / (NPAD * NPAD);
  int r = idx % (NPAD * NPAD);
  int i = r / NPAD, j = r % NPAD;
  float v;
  if (i >= NTOK || j >= NTOK) {
    v = -1e9f;
  } else {
    int ti = i / 49, hi = (i % 49) / 7, wi = i % 7;
    int tj = j / 49, hj = (j % 49) / 7, wj = j % 7;
    int id = (ti - tj + 3) * 7 + (hi - hj + 6) * 13 + (wi - wj + 6);
    v = table[id * HEADS + h];
  }
  bias224[idx] = v;
}

// ---------------- zero pad tokens 196..223 of Q, K and Vt
__global__ __launch_bounds__(256) void k_zeropad(_Float16* __restrict__ q,
                                                 _Float16* __restrict__ k,
                                                 _Float16* __restrict__ vt) {
  int idx = blockIdx.x * 256 + threadIdx.x;
  const int total = B_SZ * HEADS * 28 * HD;
  if (idx >= total) return;
  int bh = idx / (28 * HD);
  int r  = idx % (28 * HD);
  int tok = NTOK + r / HD;
  int d   = r % HD;
  size_t off = ((size_t)bh * NPAD + tok) * HD + d;
  q[off] = (_Float16)0.0f;
  k[off] = (_Float16)0.0f;
  vt[((size_t)bh * HD + d) * NPAD + tok] = (_Float16)0.0f;
}

// ---------------- QKV projection GEMM: (50176 x 384) x (384 x 1152)^T, f16 in
// 64x64 tile per 256-thread block (8 waves, each wave = 32rows x 16cols)
__global__ __launch_bounds__(256) void k_qkv(const _Float16* __restrict__ xh,
                                             const _Float16* __restrict__ wh,
                                             const float* __restrict__ bias,
                                             _Float16* __restrict__ q,
                                             _Float16* __restrict__ k,
                                             _Float16* __restrict__ vt) {
  __shared__ _Float16 sA[64 * 32];
  __shared__ _Float16 sB[64 * 32];
  const int rt = blockIdx.x, nt = blockIdx.y;
  const int t = threadIdx.x, wave = t >> 5, lane = t & 31;
  const int rowGroup = (wave >> 2) * 32;
  const int colGroup = (wave & 3) * 16;
  const int sr = t >> 2, sc = (t & 3) * 8;
  const int ml = lane & 15;
  const int ko = (lane < 16) ? 0 : 8;
  const _Float16* xg = xh + (size_t)(rt * 64 + sr) * DIMC + sc;
  const _Float16* wg = wh + (size_t)(nt * 64 + sr) * DIMC + sc;
  v8f acc0 = {}; v8f acc1 = {};
  for (int kk = 0; kk < 12; ++kk) {
    __syncthreads();
    stage16(xg + kk * 32, sA + sr * 32 + sc);
    stage16(wg + kk * 32, sB + sr * 32 + sc);
    if (kk < 11) {
      __builtin_prefetch(xg + kk * 32 + 32, 0, 0);
      __builtin_prefetch(wg + kk * 32 + 32, 0, 0);
    }
    stage_wait();
    v16h a0 = make_a_frag(sA + (rowGroup + ml) * 32, ko);
    v16h a1 = make_a_frag(sA + (rowGroup + 16 + ml) * 32, ko);
    v16h b  = *reinterpret_cast<const v16h*>(sB + (colGroup + ml) * 32 + ((lane < 16) ? 0 : 16));
    acc0 = __builtin_amdgcn_wmma_f32_16x16x32_f16(false, a0, false, b, (short)0, acc0, false, false);
    acc1 = __builtin_amdgcn_wmma_f32_16x16x32_f16(false, a1, false, b, (short)0, acc1, false, false);
  }
  const int ncol = nt * 64 + colGroup + ml;
  const int which = ncol / DIMC;
  const int h = (ncol % DIMC) / HD;
  const int d = ncol % HD;
  const float bv = bias[ncol];
#pragma unroll
  for (int tile = 0; tile < 2; ++tile) {
    v8f acc = tile ? acc1 : acc0;
#pragma unroll
    for (int r = 0; r < 8; ++r) {
      int row = rt * 64 + rowGroup + tile * 16 + r + ((lane >> 4) << 3);
      int bb = row / NTOK, tok = row % NTOK;
      size_t bh = (size_t)bb * HEADS + h;
      float val = acc[r] + bv;
      if (which == 0)      q[(bh * NPAD + tok) * HD + d] = (_Float16)(val * SCALEF);
      else if (which == 1) k[(bh * NPAD + tok) * HD + d] = (_Float16)val;
      else                 vt[(bh * HD + d) * NPAD + tok] = (_Float16)val;
    }
  }
}

// ---------------- attention: one wave per (b, h, 16-row tile)
__global__ __launch_bounds__(32) void k_attn(const _Float16* __restrict__ q,
                                             const _Float16* __restrict__ k,
                                             const _Float16* __restrict__ vt,
                                             const float* __restrict__ bias224,
                                             const float* __restrict__ mask,
                                             _Float16* __restrict__ oh) {
  __shared__ _Float16 pshare[16 * 32];
  const int rowTile = blockIdx.x;          // 0..12
  const int h = blockIdx.y;                // 0..11
  const int b = blockIdx.z;                // 0..255
  const int lane = threadIdx.x;
  const size_t bh = (size_t)b * HEADS + h;
  const _Float16* qp  = q  + bh * NPAD * HD;
  const _Float16* kp  = k  + bh * NPAD * HD;
  const _Float16* vtp = vt + bh * HD * NPAD;
  const int wmask = b & (NWIN - 1);
  const int row0 = rowTile * 16;
  const int ml = lane & 15;
  const int hi8 = (lane >> 4) << 3;        // 0 or 8
  const int ko = (lane < 16) ? 0 : 8;

  v16h aq = make_a_frag(qp + (size_t)(row0 + ml) * HD, ko);

  float sfr[14][8];
  float rmax[8];
#pragma unroll
  for (int r = 0; r < 8; ++r) rmax[r] = -3.0e38f;

  const v8f zero = {};
  const float* bph = bias224 + (size_t)h * NPAD * NPAD;
#pragma unroll
  for (int c = 0; c < 14; ++c) {
    v16h bk = *reinterpret_cast<const v16h*>(kp + (size_t)(c * 16 + ml) * HD + ((lane < 16) ? 0 : 16));
    v8f s = __builtin_amdgcn_wmma_f32_16x16x32_f16(false, aq, false, bk, (short)0, zero, false, false);
    const int j = c * 16 + ml;
#pragma unroll
    for (int r = 0; r < 8; ++r) {
      const int i = row0 + r + hi8;
      float val = s[r] + bph[(size_t)i * NPAD + j];
      if (i < NTOK && j < NTOK) val += mask[((size_t)wmask * NTOK + i) * NTOK + j];
      sfr[c][r] = val;
      rmax[r] = fmaxf(rmax[r], val);
    }
  }
#pragma unroll
  for (int off = 8; off >= 1; off >>= 1)
#pragma unroll
    for (int r = 0; r < 8; ++r)
      rmax[r] = fmaxf(rmax[r], __shfl_xor(rmax[r], off, 32));

  float rsum[8] = {0.f, 0.f, 0.f, 0.f, 0.f, 0.f, 0.f, 0.f};
#pragma unroll
  for (int c = 0; c < 14; ++c)
#pragma unroll
    for (int r = 0; r < 8; ++r) {
      float p = __expf(sfr[c][r] - rmax[r]);
      sfr[c][r] = p;
      rsum[r] += p;
    }
#pragma unroll
  for (int off = 8; off >= 1; off >>= 1)
#pragma unroll
    for (int r = 0; r < 8; ++r)
      rsum[r] += __shfl_xor(rsum[r], off, 32);

  v8f oc0 = {}; v8f oc1 = {};
#pragma unroll
  for (int s = 0; s < 7; ++s) {
#pragma unroll
    for (int tl = 0; tl < 2; ++tl) {
      const int c = 2 * s + tl;
#pragma unroll
      for (int r = 0; r < 8; ++r)
        pshare[(r + hi8) * 32 + tl * 16 + ml] = (_Float16)sfr[c][r];
    }
    __syncthreads();
    v16h pa = make_a_frag(pshare + ml * 32, ko);
    const int j0 = s * 32 + ((lane < 16) ? 0 : 16);
    v16h bv0 = *reinterpret_cast<const v16h*>(vtp + (size_t)ml * NPAD + j0);
    v16h bv1 = *reinterpret_cast<const v16h*>(vtp + (size_t)(16 + ml) * NPAD + j0);
    oc0 = __builtin_amdgcn_wmma_f32_16x16x32_f16(false, pa, false, bv0, (short)0, oc0, false, false);
    oc1 = __builtin_amdgcn_wmma_f32_16x16x32_f16(false, pa, false, bv1, (short)0, oc1, false, false);
    __syncthreads();
  }
#pragma unroll
  for (int r = 0; r < 8; ++r) {
    const int i = row0 + r + hi8;
    if (i < NTOK) {
      float inv = 1.0f / rsum[r];
      size_t base = ((size_t)b * NTOK + i) * DIMC + h * HD;
      oh[base + ml]      = (_Float16)(oc0[r] * inv);
      oh[base + 16 + ml] = (_Float16)(oc1[r] * inv);
    }
  }
}

// ---------------- output projection GEMM: (50176 x 384) x (384 x 384)^T + bias
__global__ __launch_bounds__(256) void k_proj(const _Float16* __restrict__ oh,
                                              const _Float16* __restrict__ wh,
                                              const float* __restrict__ bias,
                                              float* __restrict__ out) {
  __shared__ _Float16 sB[64 * 32];
  const int rt = blockIdx.x, nt = blockIdx.y;
  const int t = threadIdx.x, wave = t >> 5, lane = t & 31;
  const int rowGroup = (wave >> 2) * 32;
  const int colGroup = (wave & 3) * 16;
  const int sr = t >> 2, sc = (t & 3) * 8;
  const int ml = lane & 15;
  const int ko = (lane < 16) ? 0 : 8;
  const _Float16* wg = wh + (size_t)(nt * 64 + sr) * DIMC + sc;
  v8f acc0 = {}; v8f acc1 = {};
  for (int kk = 0; kk < 12; ++kk) {
    __syncthreads();
    stage16(wg + kk * 32, sB + sr * 32 + sc);
    if (kk < 11) __builtin_prefetch(wg + kk * 32 + 32, 0, 0);
    stage_wait();
    const _Float16* arow0 = oh + (size_t)(rt * 64 + rowGroup + ml) * DIMC + kk * 32;
    v16h a0 = make_a_frag(arow0, ko);
    v16h a1 = make_a_frag(arow0 + 16 * DIMC, ko);
    v16h b  = *reinterpret_cast<const v16h*>(sB + (colGroup + ml) * 32 + ((lane < 16) ? 0 : 16));
    acc0 = __builtin_amdgcn_wmma_f32_16x16x32_f16(false, a0, false, b, (short)0, acc0, false, false);
    acc1 = __builtin_amdgcn_wmma_f32_16x16x32_f16(false, a1, false, b, (short)0, acc1, false, false);
  }
  const int ncol = nt * 64 + colGroup + ml;
  const float bv = bias[ncol];
#pragma unroll
  for (int tile = 0; tile < 2; ++tile) {
    v8f acc = tile ? acc1 : acc0;
#pragma unroll
    for (int r = 0; r < 8; ++r) {
      int row = rt * 64 + rowGroup + tile * 16 + r + ((lane >> 4) << 3);
      out[(size_t)row * DIMC + ncol] = acc[r] + bv;
    }
  }
}

extern "C" void kernel_launch(void* const* d_in, const int* in_sizes, int n_in,
                              void* d_out, int out_size, void* d_ws, size_t ws_size,
                              hipStream_t stream) {
  const float* x      = (const float*)d_in[0];
  const float* mask   = (const float*)d_in[1];
  const float* qkv_w  = (const float*)d_in[2];
  const float* qkv_b  = (const float*)d_in[3];
  const float* proj_w = (const float*)d_in[4];
  const float* proj_b = (const float*)d_in[5];
  const float* table  = (const float*)d_in[6];
  float* out = (float*)d_out;

  const size_t qElems  = (size_t)B_SZ * HEADS * NPAD * HD;  // 22,020,096 halfs
  const size_t ohElems = (size_t)B_SZ * NTOK * DIMC;        // 19,267,584 halfs
  const size_t xElems  = (size_t)B_SZ * NTOK * DIMC;        // same count as oh
  const size_t wqElems = (size_t)3 * DIMC * DIMC;           // 442,368
  const size_t wpElems = (size_t)DIMC * DIMC;               // 147,456

  _Float16* qw   = (_Float16*)d_ws;
  _Float16* kw   = qw + qElems;
  _Float16* vtw  = kw + qElems;
  _Float16* ohw  = vtw + qElems;
  float*    biasw = (float*)(ohw + ohElems);
  _Float16* xhw  = (_Float16*)(biasw + (size_t)HEADS * NPAD * NPAD);
  _Float16* wqh  = xhw + xElems;
  _Float16* wph  = wqh + wqElems;

  k_cvt<<<(int)((xElems + 255) / 256), 256, 0, stream>>>(x, xhw, (int)xElems);
  k_cvt<<<(int)((wqElems + 255) / 256), 256, 0, stream>>>(qkv_w, wqh, (int)wqElems);
  k_cvt<<<(int)((wpElems + 255) / 256), 256, 0, stream>>>(proj_w, wph, (int)wpElems);
  k_bias<<<(HEADS * NPAD * NPAD + 255) / 256, 256, 0, stream>>>(table, biasw);
  k_zeropad<<<(B_SZ * HEADS * 28 * HD + 255) / 256, 256, 0, stream>>>(qw, kw, vtw);
  k_qkv<<<dim3(784, 18), 256, 0, stream>>>(xhw, wqh, qkv_b, qw, kw, vtw);
  k_attn<<<dim3(13, HEADS, B_SZ), 32, 0, stream>>>(qw, kw, vtw, biasw, mask, ohw);
  k_proj<<<dim3(784, 6), 256, 0, stream>>>(ohw, wph, proj_b, out);
}
